// QMIX_24438363914924
// MI455X (gfx1250) — matrix operand order repaired and checked by reading, not compiled, not run on hardware
//
#include <hip/hip_runtime.h>
#include <hip/hip_bf16.h>
#include <math.h>

typedef __attribute__((ext_vector_type(16))) _Float16 v16h;
typedef __attribute__((ext_vector_type(8)))  _Float16 v8h;
typedef __attribute__((ext_vector_type(8)))  float    v8f;
typedef __attribute__((ext_vector_type(4)))  float    v4f;

#define N_AGENTS   8
#define STATE_DIM  256
#define EMBED_DIM  64
#define HYPER      256
#define KS         8          // 256 / 32 k-steps
#define ACT_PITCH  272        // 256 + 16 halves pad (16B-aligned rows, rotates banks)
#define F32_PITCH  68         // 64 + 4 floats pad
#define ROWS       32         // rows per wave (2 M-tiles)

union V16U { v16h v; v8h h[2]; };

// ---------------------------------------------------------------------------
// Prep: repack f32 weight W[K][N] (row-major) into f16 WMMA B-fragments:
//   out[(((ntile*KS + kstep)*32 + lane)*16 + e]
//     = W[kstep*32 + (e>>3)*16 + (lane>=16)*8 + (e&7)][ntile*16 + (lane&15)]
// (16-bit B 32x16 layout: lanes 0-15 hold K=0..7,16..23, lanes 16-31 hold K=8..15,24..31)
// ---------------------------------------------------------------------------
__global__ void qmix_pack_w(const float* __restrict__ W, _Float16* __restrict__ out,
                            int K, int N) {
    int i = blockIdx.x * blockDim.x + threadIdx.x;
    int total = K * N;
    if (i >= total) return;
    int e     = i & 15;
    int lane  = (i >> 4) & 31;
    int frag  = i >> 9;
    int ksteps = K >> 5;
    int kstep = frag % ksteps;
    int ntile = frag / ksteps;
    int k   = kstep * 32 + ((e >> 3) << 4) + ((lane >> 4) << 3) + (e & 7);
    int col = ntile * 16 + (lane & 15);
    out[i] = (_Float16)W[k * N + col];
}

// ---------------------------------------------------------------------------
// Fragment loaders
// ---------------------------------------------------------------------------
__device__ inline v16h load_frag32B(const _Float16* p) {
    V16U u;
    u.h[0] = *(const v8h*)(p);
    u.h[1] = *(const v8h*)(p + 8);
    return u.v;
}

// A-fragment from LDS activation buffer, row-major [ROWS][ACT_PITCH] f16.
// 16-bit A 16x32: row = mtile*16 + (lane&15);
//                 k = ks*32 + (e>>3)*16 + (lane>=16)*8 + (e&7)
__device__ inline v16h load_afrag_act(const _Float16* act, int ks, int lane, int mtile) {
    int row  = mtile * 16 + (lane & 15);
    int half = lane >> 4;
    const _Float16* p = act + row * ACT_PITCH + ks * 32 + half * 8;
    V16U u;
    u.h[0] = *(const v8h*)(p);        // K = base .. base+7
    u.h[1] = *(const v8h*)(p + 16);   // K = base+16 .. base+23
    return u.v;
}

__device__ inline v8f wmma_f16(v16h a, v16h b, v8f c) {
    return __builtin_amdgcn_wmma_f32_16x16x32_f16(false, a, false, b,
                                                  (short)0, c, false, false);
}

struct Acc22 { v8f c00, c01, c10, c11; };

// 2x2 register-blocked GEMM: A fragments from pre-fragmented LDS X buffer.
// Per k-step: 2 LDS loads + 2 global loads feed 4 WMMAs.
__device__ inline Acc22 gemm22_frag(const _Float16* xf0, const _Float16* xf1,
                                    const _Float16* __restrict__ wfrag,
                                    int t0, int lane) {
    Acc22 o; o.c00 = {}; o.c01 = {}; o.c10 = {}; o.c11 = {};
#pragma unroll
    for (int ks = 0; ks < KS; ++ks) {
        v16h a0 = load_frag32B(xf0 + (ks * 32 + lane) * 16);
        v16h a1 = load_frag32B(xf1 + (ks * 32 + lane) * 16);
        v16h b0 = load_frag32B(wfrag + ((((t0    ) * KS + ks) * 32) + lane) * 16);
        v16h b1 = load_frag32B(wfrag + ((((t0 + 1) * KS + ks) * 32) + lane) * 16);
        o.c00 = wmma_f16(a0, b0, o.c00);
        o.c01 = wmma_f16(a0, b1, o.c01);
        o.c10 = wmma_f16(a1, b0, o.c10);
        o.c11 = wmma_f16(a1, b1, o.c11);
    }
    return o;
}

// Same, A from the row-major padded LDS activation buffer (layer 2).
__device__ inline Acc22 gemm22_act(const _Float16* act,
                                   const _Float16* __restrict__ wfrag,
                                   int t0, int lane) {
    Acc22 o; o.c00 = {}; o.c01 = {}; o.c10 = {}; o.c11 = {};
#pragma unroll
    for (int ks = 0; ks < KS; ++ks) {
        v16h a0 = load_afrag_act(act, ks, lane, 0);
        v16h a1 = load_afrag_act(act, ks, lane, 1);
        v16h b0 = load_frag32B(wfrag + ((((t0    ) * KS + ks) * 32) + lane) * 16);
        v16h b1 = load_frag32B(wfrag + ((((t0 + 1) * KS + ks) * 32) + lane) * 16);
        o.c00 = wmma_f16(a0, b0, o.c00);
        o.c01 = wmma_f16(a0, b1, o.c01);
        o.c10 = wmma_f16(a1, b0, o.c10);
        o.c11 = wmma_f16(a1, b1, o.c11);
    }
    return o;
}

// ---------------------------------------------------------------------------
// Fused QMIX mixer. One wave per block, 32 batch rows per wave, 2x2 blocking.
// ---------------------------------------------------------------------------
__global__ __launch_bounds__(32) void qmix_main(
    const float* __restrict__ agent_qs,   // [B, 8]
    const float* __restrict__ states,     // [B, 256]
    const _Float16* __restrict__ fw1a,    // fragment-packed weights (f16)
    const _Float16* __restrict__ fw1b,
    const _Float16* __restrict__ fwfa,
    const _Float16* __restrict__ fwfb,
    const _Float16* __restrict__ fwb1,
    const _Float16* __restrict__ fwva,
    const float* __restrict__ b1a, const float* __restrict__ b1b,
    const float* __restrict__ bfa, const float* __restrict__ bfb,
    const float* __restrict__ bb1, const float* __restrict__ bva,
    const float* __restrict__ wvb, const float* __restrict__ bvb,
    float* __restrict__ out)
{
    __shared__ __align__(16) _Float16 s_xfrag[2][KS * 32 * 16];   // 16 KB, per M-tile
    __shared__ __align__(16) _Float16 s_act[ROWS * ACT_PITCH];    // 17 KB
    __shared__ __align__(16) float    s_b1[ROWS * F32_PITCH];     // hidden accumulator
    __shared__ __align__(16) float    s_hv[ROWS * F32_PITCH];
    __shared__ __align__(16) float    s_wf[ROWS * F32_PITCH];
    __shared__ __align__(16) float    s_qs[ROWS * N_AGENTS];

    const int lane = threadIdx.x & 31;
    const int half = lane >> 4;
    const int col  = lane & 15;
    const long rowBase = (long)blockIdx.x * ROWS;

    // ---- Phase 0: stage X (f32 -> f16, A-fragment order) + agent_qs ----
#pragma unroll
    for (int m = 0; m < 2; ++m) {
        const float* Xr = states + (rowBase + m * 16 + col) * STATE_DIM;
#pragma unroll
        for (int ks = 0; ks < KS; ++ks) {
            int k0 = ks * 32 + half * 8;
            v4f x0 = *(const v4f*)(Xr + k0);
            v4f x1 = *(const v4f*)(Xr + k0 + 4);
            v4f x2 = *(const v4f*)(Xr + k0 + 16);
            v4f x3 = *(const v4f*)(Xr + k0 + 20);
            _Float16* d = s_xfrag[m] + (ks * 32 + lane) * 16;
#pragma unroll
            for (int j = 0; j < 4; ++j) {
                d[j]      = (_Float16)x0[j];
                d[4 + j]  = (_Float16)x1[j];
                d[8 + j]  = (_Float16)x2[j];
                d[12 + j] = (_Float16)x3[j];
            }
        }
    }
    {   // one row of agent_qs per lane
        v4f q0 = *(const v4f*)(agent_qs + (rowBase + lane) * N_AGENTS);
        v4f q1 = *(const v4f*)(agent_qs + (rowBase + lane) * N_AGENTS + 4);
        *(v4f*)&s_qs[lane * N_AGENTS]     = q0;
        *(v4f*)&s_qs[lane * N_AGENTS + 4] = q1;
    }
    __syncthreads();

    // ---- Phase 1a: b1 = states@wb1 + bb1 -> hidden accumulator base ----
#pragma unroll
    for (int tp = 0; tp < (EMBED_DIM / 16) / 2; ++tp) {
        int t0 = 2 * tp;
        Acc22 o = gemm22_frag(s_xfrag[0], s_xfrag[1], fwb1, t0, lane);
        float bias0 = bb1[t0 * 16 + col];
        float bias1 = bb1[(t0 + 1) * 16 + col];
#pragma unroll
        for (int r = 0; r < 8; ++r) {
            int r0 = r + half * 8, r1 = 16 + r + half * 8;
            s_b1[r0 * F32_PITCH + t0 * 16 + col]       = o.c00[r] + bias0;
            s_b1[r0 * F32_PITCH + (t0 + 1) * 16 + col] = o.c01[r] + bias1;
            s_b1[r1 * F32_PITCH + t0 * 16 + col]       = o.c10[r] + bias0;
            s_b1[r1 * F32_PITCH + (t0 + 1) * 16 + col] = o.c11[r] + bias1;
        }
    }

    // ---- Phase 1b: hv = relu(states@wva + bva) ----
#pragma unroll
    for (int tp = 0; tp < (EMBED_DIM / 16) / 2; ++tp) {
        int t0 = 2 * tp;
        Acc22 o = gemm22_frag(s_xfrag[0], s_xfrag[1], fwva, t0, lane);
        float bias0 = bva[t0 * 16 + col];
        float bias1 = bva[(t0 + 1) * 16 + col];
#pragma unroll
        for (int r = 0; r < 8; ++r) {
            int r0 = r + half * 8, r1 = 16 + r + half * 8;
            float h00 = o.c00[r] + bias0, h01 = o.c01[r] + bias1;
            float h10 = o.c10[r] + bias0, h11 = o.c11[r] + bias1;
            s_hv[r0 * F32_PITCH + t0 * 16 + col]       = h00 > 0.f ? h00 : 0.f;
            s_hv[r0 * F32_PITCH + (t0 + 1) * 16 + col] = h01 > 0.f ? h01 : 0.f;
            s_hv[r1 * F32_PITCH + t0 * 16 + col]       = h10 > 0.f ? h10 : 0.f;
            s_hv[r1 * F32_PITCH + (t0 + 1) * 16 + col] = h11 > 0.f ? h11 : 0.f;
        }
    }

    // ---- Phase 1c: h1 = relu(states@w1a + b1a) -> f16 act buffer ----
    for (int tp = 0; tp < (HYPER / 16) / 2; ++tp) {
        int t0 = 2 * tp;
        Acc22 o = gemm22_frag(s_xfrag[0], s_xfrag[1], fw1a, t0, lane);
        float bias0 = b1a[t0 * 16 + col];
        float bias1 = b1a[(t0 + 1) * 16 + col];
#pragma unroll
        for (int r = 0; r < 8; ++r) {
            int r0 = r + half * 8, r1 = 16 + r + half * 8;
            float h00 = o.c00[r] + bias0, h01 = o.c01[r] + bias1;
            float h10 = o.c10[r] + bias0, h11 = o.c11[r] + bias1;
            s_act[r0 * ACT_PITCH + t0 * 16 + col]       = (_Float16)(h00 > 0.f ? h00 : 0.f);
            s_act[r0 * ACT_PITCH + (t0 + 1) * 16 + col] = (_Float16)(h01 > 0.f ? h01 : 0.f);
            s_act[r1 * ACT_PITCH + t0 * 16 + col]       = (_Float16)(h10 > 0.f ? h10 : 0.f);
            s_act[r1 * ACT_PITCH + (t0 + 1) * 16 + col] = (_Float16)(h11 > 0.f ? h11 : 0.f);
        }
    }
    __syncthreads();

    // ---- Phase 2: w1 = |h1@w1b + b1b|, fused agent mixing into s_b1 ----
    for (int tp = 0; tp < ((EMBED_DIM * N_AGENTS) / 16) / 2; ++tp) {
        int t0 = 2 * tp;
        Acc22 o = gemm22_act(s_act, fw1b, t0, lane);
        int gc0 = t0 * 16 + col, gc1 = (t0 + 1) * 16 + col;
        float bias0 = b1b[gc0], bias1 = b1b[gc1];
        int a0 = gc0 >> 6, e0 = gc0 & 63;
        int a1 = gc1 >> 6, e1 = gc1 & 63;
#pragma unroll
        for (int r = 0; r < 8; ++r) {
            int r0 = r + half * 8, r1 = 16 + r + half * 8;
            atomicAdd(&s_b1[r0 * F32_PITCH + e0],
                      s_qs[r0 * N_AGENTS + a0] * fabsf(o.c00[r] + bias0));
            atomicAdd(&s_b1[r0 * F32_PITCH + e1],
                      s_qs[r0 * N_AGENTS + a1] * fabsf(o.c01[r] + bias1));
            atomicAdd(&s_b1[r1 * F32_PITCH + e0],
                      s_qs[r1 * N_AGENTS + a0] * fabsf(o.c10[r] + bias0));
            atomicAdd(&s_b1[r1 * F32_PITCH + e1],
                      s_qs[r1 * N_AGENTS + a1] * fabsf(o.c11[r] + bias1));
        }
    }
    __syncthreads();

    // ---- Phase 3: hf = relu(states@wfa + bfa) -> reuse act buffer ----
    for (int tp = 0; tp < (HYPER / 16) / 2; ++tp) {
        int t0 = 2 * tp;
        Acc22 o = gemm22_frag(s_xfrag[0], s_xfrag[1], fwfa, t0, lane);
        float bias0 = bfa[t0 * 16 + col];
        float bias1 = bfa[(t0 + 1) * 16 + col];
#pragma unroll
        for (int r = 0; r < 8; ++r) {
            int r0 = r + half * 8, r1 = 16 + r + half * 8;
            float h00 = o.c00[r] + bias0, h01 = o.c01[r] + bias1;
            float h10 = o.c10[r] + bias0, h11 = o.c11[r] + bias1;
            s_act[r0 * ACT_PITCH + t0 * 16 + col]       = (_Float16)(h00 > 0.f ? h00 : 0.f);
            s_act[r0 * ACT_PITCH + (t0 + 1) * 16 + col] = (_Float16)(h01 > 0.f ? h01 : 0.f);
            s_act[r1 * ACT_PITCH + t0 * 16 + col]       = (_Float16)(h10 > 0.f ? h10 : 0.f);
            s_act[r1 * ACT_PITCH + (t0 + 1) * 16 + col] = (_Float16)(h11 > 0.f ? h11 : 0.f);
        }
    }
    __syncthreads();

    // ---- Phase 4: w_final = |hf@wfb + bfb| ----
#pragma unroll
    for (int tp = 0; tp < (EMBED_DIM / 16) / 2; ++tp) {
        int t0 = 2 * tp;
        Acc22 o = gemm22_act(s_act, fwfb, t0, lane);
        float bias0 = bfb[t0 * 16 + col];
        float bias1 = bfb[(t0 + 1) * 16 + col];
#pragma unroll
        for (int r = 0; r < 8; ++r) {
            int r0 = r + half * 8, r1 = 16 + r + half * 8;
            s_wf[r0 * F32_PITCH + t0 * 16 + col]       = fabsf(o.c00[r] + bias0);
            s_wf[r0 * F32_PITCH + (t0 + 1) * 16 + col] = fabsf(o.c01[r] + bias1);
            s_wf[r1 * F32_PITCH + t0 * 16 + col]       = fabsf(o.c10[r] + bias0);
            s_wf[r1 * F32_PITCH + (t0 + 1) * 16 + col] = fabsf(o.c11[r] + bias1);
        }
    }
    __syncthreads();

    // ---- Phase 5: epilogue (VALU), one row per lane ----
    {
        int row = lane;
        float v = bvb[0];
#pragma unroll 8
        for (int k = 0; k < EMBED_DIM; ++k)
            v += s_hv[row * F32_PITCH + k] * wvb[k];
        float y = v;
#pragma unroll 8
        for (int e = 0; e < EMBED_DIM; ++e) {
            float h = s_b1[row * F32_PITCH + e];
            h = h > 0.f ? h : (__expf(h) - 1.f);           // ELU(alpha=1)
            y += h * s_wf[row * F32_PITCH + e];
        }
        out[rowBase + row] = y;
    }
}

// ---------------------------------------------------------------------------
// Host launch
// ---------------------------------------------------------------------------
extern "C" void kernel_launch(void* const* d_in, const int* in_sizes, int n_in,
                              void* d_out, int out_size, void* d_ws, size_t ws_size,
                              hipStream_t stream) {
    const float* agent_qs = (const float*)d_in[0];
    const float* states   = (const float*)d_in[1];
    const float* w1a = (const float*)d_in[2];
    const float* b1a = (const float*)d_in[3];
    const float* w1b = (const float*)d_in[4];
    const float* b1b = (const float*)d_in[5];
    const float* wfa = (const float*)d_in[6];
    const float* bfa = (const float*)d_in[7];
    const float* wfb = (const float*)d_in[8];
    const float* bfb = (const float*)d_in[9];
    const float* wb1 = (const float*)d_in[10];
    const float* bb1 = (const float*)d_in[11];
    const float* wva = (const float*)d_in[12];
    const float* bva = (const float*)d_in[13];
    const float* wvb = (const float*)d_in[14];
    const float* bvb = (const float*)d_in[15];

    const int B = in_sizes[0] / N_AGENTS;

    // Workspace layout (f16 fragment-packed weights)
    _Float16* ws = (_Float16*)d_ws;
    _Float16* fw1a = ws;                       // 256*256
    _Float16* fw1b = fw1a + 256 * 256;         // 256*512
    _Float16* fwfa = fw1b + 256 * 512;         // 256*256
    _Float16* fwfb = fwfa + 256 * 256;         // 256*64
    _Float16* fwb1 = fwfb + 256 * 64;          // 256*64
    _Float16* fwva = fwb1 + 256 * 64;          // 256*64

    const int TB = 256;
    qmix_pack_w<<<(256 * 256 + TB - 1) / TB, TB, 0, stream>>>(w1a, fw1a, 256, 256);
    qmix_pack_w<<<(256 * 512 + TB - 1) / TB, TB, 0, stream>>>(w1b, fw1b, 256, 512);
    qmix_pack_w<<<(256 * 256 + TB - 1) / TB, TB, 0, stream>>>(wfa, fwfa, 256, 256);
    qmix_pack_w<<<(256 * 64  + TB - 1) / TB, TB, 0, stream>>>(wfb, fwfb, 256, 64);
    qmix_pack_w<<<(256 * 64  + TB - 1) / TB, TB, 0, stream>>>(wb1, fwb1, 256, 64);
    qmix_pack_w<<<(256 * 64  + TB - 1) / TB, TB, 0, stream>>>(wva, fwva, 256, 64);

    // 1 wave/block, 32 rows per wave
    dim3 grid(B / ROWS), block(32);
    qmix_main<<<grid, block, 0, stream>>>(
        agent_qs, states,
        fw1a, fw1b, fwfa, fwfb, fwb1, fwva,
        b1a, b1b, bfa, bfb, bb1, bva, wvb, bvb,
        (float*)d_out);
}
